// ImageVQEncoder_31129922961671
// MI455X (gfx1250) — matrix operand order repaired and checked by reading, not compiled, not run on hardware
//
#include <hip/hip_runtime.h>

// ---------------- WMMA fragment types (CDNA5, wave32) ----------------
typedef __attribute__((ext_vector_type(16))) __bf16 v16bf;
typedef __attribute__((ext_vector_type(8)))  __bf16 v8bf;
typedef __attribute__((ext_vector_type(8)))  float  v8f;

// A fragment: 16x32 bf16. Lane L -> row M = L&15, half = L>>4.
// Code's k order (B matches): lane loads K = half*16 + [0..7] (VGPR0-3) and K = half*16 + 8 + [0..7] (VGPR4-7):
// two contiguous 16-byte runs when K (channels) is contiguous in memory.
__device__ inline v16bf load_a_frag(const __bf16* rowp, int half) {
  v8bf lo = *(const v8bf*)(rowp + half * 16);
  v8bf hi = *(const v8bf*)(rowp + half * 16 + 8);
  v16bf a;
#pragma unroll
  for (int j = 0; j < 8; ++j) { a[j] = lo[j]; a[8 + j] = hi[j]; }
  return a;
}

__device__ inline v16bf zero_a_frag() {
  v16bf a;
#pragma unroll
  for (int j = 0; j < 16; ++j) a[j] = (__bf16)0.0f;
  return a;
}

// B fragment: 32x16 bf16. Lane L -> col N = L&15, K = (L>>4)*16 + [0..15]:
// one contiguous 32-byte run when stored as [N][K].

__device__ inline v8f wmma_bf16(const v16bf a, const v16bf b, const v8f c) {
  return __builtin_amdgcn_wmma_f32_16x16x32_bf16(false, a, false, b,
                                                 (short)0, c, false, false);
}

// order-preserving float->uint map packed with the column index
__device__ inline unsigned long long make_key(float score, unsigned int col) {
  unsigned int ub = __float_as_uint(score);
  ub = (ub & 0x80000000u) ? ~ub : (ub | 0x80000000u);
  return ((unsigned long long)ub << 32) | col;
}

// min-reduce a u64 key across each 16-lane group
__device__ inline unsigned long long shflxor_min16(unsigned long long key) {
#pragma unroll
  for (int mask = 1; mask < 16; mask <<= 1) {
    unsigned int lo = (unsigned int)key, hi = (unsigned int)(key >> 32);
    lo = (unsigned int)__shfl_xor((int)lo, mask, 16);
    hi = (unsigned int)__shfl_xor((int)hi, mask, 16);
    const unsigned long long other = ((unsigned long long)hi << 32) | lo;
    if (other < key) key = other;
  }
  return key;
}

// ---------------- conv1 (3->128): direct, tiny K, 2.4% of FLOPs ----------------
__global__ __launch_bounds__(256) void conv1_direct(
    const float* __restrict__ img, const float* __restrict__ w1,
    const float* __restrict__ b1, __bf16* __restrict__ out) {
  // out NHWC [32][128][128][128]
  size_t idx = (size_t)blockIdx.x * 256 + threadIdx.x;
  const int c = (int)(idx & 127);
  const int q = (int)((idx >> 7) & 127);
  const int p = (int)((idx >> 14) & 127);
  const int b = (int)(idx >> 21);
  float acc = b1[c];
#pragma unroll
  for (int ci = 0; ci < 3; ++ci) {
#pragma unroll
    for (int kh = 0; kh < 4; ++kh) {
      const int y = 2 * p - 1 + kh;
      if ((unsigned)y >= 256u) continue;
#pragma unroll
      for (int kw = 0; kw < 4; ++kw) {
        const int x = 2 * q - 1 + kw;
        if ((unsigned)x >= 256u) continue;
        acc += img[(((size_t)b * 3 + ci) * 256 + y) * 256 + x] *
               w1[(size_t)(c * 3 + ci) * 16 + kh * 4 + kw];
      }
    }
  }
  const float s = acc / (1.0f + __expf(-acc));
  out[idx] = (__bf16)s;
}

// ---------------- weight transform: f32 OIHW -> bf16 [tap][O][C] ----------------
__global__ __launch_bounds__(256) void xform_w(const float* __restrict__ w,
                                               __bf16* __restrict__ wT,
                                               int O, int C) {
  size_t idx = (size_t)blockIdx.x * 256 + threadIdx.x;
  const size_t total = (size_t)O * C * 16;
  if (idx >= total) return;
  const int tap = (int)(idx & 15);
  const size_t oc = idx >> 4;
  const int c = (int)(oc % C);
  const int o = (int)(oc / C);
  wT[((size_t)tap * O + o) * C + c] = (__bf16)w[oc * 16 + tap];
}

// ---------------- codebook: bf16 cast + per-row squared norm ----------------
__global__ __launch_bounds__(256) void xform_cb(const float* __restrict__ cb,
                                                __bf16* __restrict__ cbb,
                                                float* __restrict__ cnorm) {
  const int k = blockIdx.x;
  const float v = cb[(size_t)k * 256 + threadIdx.x];
  cbb[(size_t)k * 256 + threadIdx.x] = (__bf16)v;
  float s = v * v;
#pragma unroll
  for (int off = 16; off > 0; off >>= 1) s += __shfl_down(s, off, 32);
  __shared__ float sm[8];
  if ((threadIdx.x & 31) == 0) sm[threadIdx.x >> 5] = s;
  __syncthreads();
  if (threadIdx.x == 0) {
    float t = 0.f;
#pragma unroll
    for (int i = 0; i < 8; ++i) t += sm[i];
    cnorm[k] = t;
  }
}

// ------- implicit-GEMM conv + SiLU, 32x32 tile/wave (2x2 WMMA blocking) --------
// M = 32 flattened output pixels, N = 32 output channels, K = CIN*16 taps.
template <int CIN>
__global__ __launch_bounds__(256) void conv_silu_wmma(
    const __bf16* __restrict__ in, const __bf16* __restrict__ wT,
    const float* __restrict__ bias, __bf16* __restrict__ out,
    float* __restrict__ out_f32,
    int B, int Hin, int Win, int O, int Ho, int Wo) {
  const int lane = threadIdx.x & 31;
  const int m    = lane & 15;
  const int half = lane >> 4;
  long long wid = (long long)blockIdx.x * 8 + (threadIdx.x >> 5);
  const int nNblk   = O >> 5;
  const int nPixBlk = (Ho * Wo) >> 5;
  const int nb = (int)(wid % nNblk);   wid /= nNblk;
  const int pb = (int)(wid % nPixBlk); wid /= nPixBlk;
  const int b  = (int)wid;
  if (b >= B) return;

  const int o0  = nb << 5;
  const int hw0 = pb << 5;
  // this lane's two A rows (flattened output pixels)
  const int pix0 = hw0 + m, pix1 = hw0 + 16 + m;
  const int p0 = pix0 / Wo, q0 = pix0 % Wo;
  const int p1 = pix1 / Wo, q1 = pix1 % Wo;

  v8f acc00 = {}, acc01 = {}, acc10 = {}, acc11 = {};
#pragma unroll 1
  for (int tap = 0; tap < 16; ++tap) {
    const int kh = tap >> 2, kw = tap & 3;
    const int y0 = 2 * p0 - 1 + kh, x0 = 2 * q0 - 1 + kw;
    const int y1 = 2 * p1 - 1 + kh, x1 = 2 * q1 - 1 + kw;
    const bool v0 = ((unsigned)y0 < (unsigned)Hin) && ((unsigned)x0 < (unsigned)Win);
    const bool v1 = ((unsigned)y1 < (unsigned)Hin) && ((unsigned)x1 < (unsigned)Win);
    const __bf16* ap0 = in + (((size_t)b * Hin + y0) * Win + x0) * CIN;
    const __bf16* ap1 = in + (((size_t)b * Hin + y1) * Win + x1) * CIN;
    const __bf16* wp0 = wT + ((size_t)tap * O + o0 + m) * CIN + half * 16;
    const __bf16* wp1 = wp0 + (size_t)16 * CIN;
    if (tap < 15)  // stream next tap's weights (global_prefetch_b8)
      __builtin_prefetch(wT + ((size_t)(tap + 1) * O + o0 + m) * CIN + half * 16, 0, 3);
#pragma unroll
    for (int c0 = 0; c0 < CIN; c0 += 32) {
      const v16bf a0 = v0 ? load_a_frag(ap0 + c0, half) : zero_a_frag();
      const v16bf a1 = v1 ? load_a_frag(ap1 + c0, half) : zero_a_frag();
      const v16bf b0 = *(const v16bf*)(wp0 + c0);
      const v16bf b1 = *(const v16bf*)(wp1 + c0);
      acc00 = wmma_bf16(a0, b0, acc00);
      acc01 = wmma_bf16(a0, b1, acc01);
      acc10 = wmma_bf16(a1, b0, acc10);
      acc11 = wmma_bf16(a1, b1, acc11);
    }
  }
  const float bv0 = bias[o0 + m];
  const float bv1 = bias[o0 + 16 + m];
  const size_t bbase = (size_t)b * Ho * Wo;
  auto store_tile = [&](const v8f& acc, int subm, int subn, float bv) {
#pragma unroll
    for (int i = 0; i < 8; ++i) {
      const int pix = hw0 + subm * 16 + half * 8 + i;  // C-tile row map
      const float yv = acc[i] + bv;
      const float s  = yv / (1.0f + __expf(-yv));
      const size_t oidx = (bbase + pix) * O + o0 + subn * 16 + m;
      out[oidx] = (__bf16)s;
      if (out_f32) out_f32[oidx] = s;
    }
  };
  store_tile(acc00, 0, 0, bv0);
  store_tile(acc01, 0, 1, bv1);
  store_tile(acc10, 1, 0, bv0);
  store_tile(acc11, 1, 1, bv1);
}

// -------- VQ: 32x32 WMMA tiles, score = |c|^2 - 2 z.c, hierarchical argmin -----
__global__ void init_keys(unsigned long long* __restrict__ keys) {
  const int i = blockIdx.x * 256 + threadIdx.x;
  if (i < 8192) keys[i] = ~0ull;
}

// Block = 32 z-rows x 256 codes (8 waves, each a 32x32 tile).
// LDS u64 atomicMin combines waves, then 32 global atomicMin per block.
__global__ __launch_bounds__(256) void vq_argmin_wmma(
    const __bf16* __restrict__ zb, const __bf16* __restrict__ cbb,
    const float* __restrict__ cnorm, unsigned long long* __restrict__ keys) {
  const int lane = threadIdx.x & 31;
  const int wv   = threadIdx.x >> 5;
  const int m = lane & 15, half = lane >> 4;
  const int nColGrp = 8192 / 256;                  // 32 column groups
  const int r0 = (blockIdx.x / nColGrp) * 32;
  const int n0 = (blockIdx.x % nColGrp) * 256 + wv * 32;

  __shared__ unsigned long long skeys[32];
  if (threadIdx.x < 32) skeys[threadIdx.x] = ~0ull;
  __syncthreads();

  const __bf16* ap0 = zb  + (size_t)(r0 + m) * 256;
  const __bf16* ap1 = zb  + (size_t)(r0 + 16 + m) * 256;
  const __bf16* bp0 = cbb + (size_t)(n0 + m) * 256 + half * 16;
  const __bf16* bp1 = cbb + (size_t)(n0 + 16 + m) * 256 + half * 16;
  v8f acc00 = {}, acc01 = {}, acc10 = {}, acc11 = {};
#pragma unroll
  for (int k = 0; k < 256; k += 32) {
    const v16bf a0 = load_a_frag(ap0 + k, half);
    const v16bf a1 = load_a_frag(ap1 + k, half);
    const v16bf b0 = *(const v16bf*)(bp0 + k);
    const v16bf b1 = *(const v16bf*)(bp1 + k);
    acc00 = wmma_bf16(a0, b0, acc00);
    acc01 = wmma_bf16(a0, b1, acc01);
    acc10 = wmma_bf16(a1, b0, acc10);
    acc11 = wmma_bf16(a1, b1, acc11);
  }
  const float cn0 = cnorm[n0 + m];
  const float cn1 = cnorm[n0 + 16 + m];
  const unsigned int col0 = (unsigned int)(n0 + m);
  const unsigned int col1 = (unsigned int)(n0 + 16 + m);
  auto reduce_rows = [&](const v8f& ac0, const v8f& ac1, int rowbase) {
#pragma unroll
    for (int i = 0; i < 8; ++i) {
      unsigned long long key = make_key(cn0 - 2.0f * ac0[i], col0);
      const unsigned long long k1 = make_key(cn1 - 2.0f * ac1[i], col1);
      if (k1 < key) key = k1;
      key = shflxor_min16(key);
      if (m == 0) atomicMin(&skeys[rowbase + half * 8 + i], key);  // ds_min path
    }
  };
  reduce_rows(acc00, acc01, 0);
  reduce_rows(acc10, acc11, 16);
  __syncthreads();
  if (threadIdx.x < 32)
    atomicMin(&keys[r0 + threadIdx.x], skeys[threadIdx.x]);  // deterministic (min)
}

__global__ __launch_bounds__(256) void vq_finalize(
    const unsigned long long* __restrict__ keys, const float* __restrict__ cb,
    const float* __restrict__ z4f, float* __restrict__ qout,
    float* __restrict__ idxout, float* __restrict__ partials) {
  const int r = blockIdx.x;          // r = b*256 + hw
  const int d = threadIdx.x;
  const unsigned int idx = (unsigned int)(keys[r] & 0xFFFFFFFFull);
  const int b = r >> 8, hw = r & 255;
  const float qv = cb[(size_t)idx * 256 + d];
  qout[((size_t)b * 256 + d) * 256 + hw] = qv;  // NCHW output
  const float z = z4f[(size_t)r * 256 + d];
  float s = (z - qv) * (z - qv);
#pragma unroll
  for (int off = 16; off > 0; off >>= 1) s += __shfl_down(s, off, 32);
  __shared__ float sm[8];
  if ((threadIdx.x & 31) == 0) sm[threadIdx.x >> 5] = s;
  __syncthreads();
  if (threadIdx.x == 0) {
    float t = 0.f;
#pragma unroll
    for (int i = 0; i < 8; ++i) t += sm[i];
    partials[r] = t;
    idxout[r] = (float)idx;
  }
}

__global__ __launch_bounds__(256) void reduce_loss(
    const float* __restrict__ partials, float* __restrict__ loss) {
  float s = 0.f;
  for (int i = threadIdx.x; i < 8192; i += 256) s += partials[i];
#pragma unroll
  for (int off = 16; off > 0; off >>= 1) s += __shfl_down(s, off, 32);
  __shared__ float sm[8];
  if ((threadIdx.x & 31) == 0) sm[threadIdx.x >> 5] = s;
  __syncthreads();
  if (threadIdx.x == 0) {
    float t = 0.f;
#pragma unroll
    for (int i = 0; i < 8; ++i) t += sm[i];
    *loss = t * (1.0f / 2097152.0f);  // mean over 32*256*16*16
  }
}

// ---------------- host launch ----------------
extern "C" void kernel_launch(void* const* d_in, const int* in_sizes, int n_in,
                              void* d_out, int out_size, void* d_ws, size_t ws_size,
                              hipStream_t stream) {
  const float* images = (const float*)d_in[0];
  const float* w1 = (const float*)d_in[1];
  const float* b1 = (const float*)d_in[2];
  const float* w2 = (const float*)d_in[3];
  const float* b2 = (const float*)d_in[4];
  const float* w3 = (const float*)d_in[5];
  const float* b3 = (const float*)d_in[6];
  const float* w4 = (const float*)d_in[7];
  const float* b4 = (const float*)d_in[8];
  const float* cbf = (const float*)d_in[9];

  char* ws = (char*)d_ws;
  size_t off = 0;
  auto take = [&](size_t bytes) -> char* {
    char* p = ws + off;
    off += (bytes + 255) & ~(size_t)255;
    return p;
  };
  __bf16* z1  = (__bf16*)take((size_t)32 * 128 * 128 * 128 * 2);  // NHWC
  __bf16* z2  = (__bf16*)take((size_t)32 * 64 * 64 * 256 * 2);
  __bf16* z3  = (__bf16*)take((size_t)32 * 32 * 32 * 256 * 2);
  __bf16* z4b = (__bf16*)take((size_t)32 * 16 * 16 * 256 * 2);
  float*  z4f = (float*) take((size_t)32 * 16 * 16 * 256 * 4);
  __bf16* w2t = (__bf16*)take((size_t)16 * 256 * 128 * 2);
  __bf16* w3t = (__bf16*)take((size_t)16 * 256 * 256 * 2);
  __bf16* w4t = (__bf16*)take((size_t)16 * 256 * 256 * 2);
  __bf16* cbb = (__bf16*)take((size_t)8192 * 256 * 2);
  float*  cnorm = (float*)take((size_t)8192 * 4);
  unsigned long long* keys = (unsigned long long*)take((size_t)8192 * 8);
  float* partials = (float*)take((size_t)8192 * 4);

  float* qout   = (float*)d_out;       // [32,256,16,16] f32
  float* idxout = qout + 2097152;      // [32,256] (indices as float)
  float* lossp  = idxout + 8192;       // scalar

  init_keys<<<32, 256, 0, stream>>>(keys);
  xform_w<<<(16 * 256 * 128 + 255) / 256, 256, 0, stream>>>(w2, w2t, 256, 128);
  xform_w<<<(16 * 256 * 256 + 255) / 256, 256, 0, stream>>>(w3, w3t, 256, 256);
  xform_w<<<(16 * 256 * 256 + 255) / 256, 256, 0, stream>>>(w4, w4t, 256, 256);
  xform_cb<<<8192, 256, 0, stream>>>(cbf, cbb, cnorm);
  conv1_direct<<<262144, 256, 0, stream>>>(images, w1, b1, z1);

  // conv2: z1[32,128,128,128] -> z2[32,64,64,256]; waves = 32*128*8
  conv_silu_wmma<128><<<(32 * 128 * 8) / 8, 256, 0, stream>>>(
      z1, w2t, b2, z2, nullptr, 32, 128, 128, 256, 64, 64);
  // conv3: z2 -> z3[32,32,32,256]; waves = 32*32*8
  conv_silu_wmma<256><<<(32 * 32 * 8) / 8, 256, 0, stream>>>(
      z2, w3t, b3, z3, nullptr, 32, 64, 64, 256, 32, 32);
  // conv4: z3 -> z4 (bf16 for VQ GEMM, f32 for loss); waves = 32*8*8
  conv_silu_wmma<256><<<(32 * 8 * 8) / 8, 256, 0, stream>>>(
      z3, w4t, b4, z4b, z4f, 32, 32, 32, 256, 16, 16);

  // VQ: 256 row-blocks x 32 col-groups; 8 waves/block each own a 32x32 tile
  vq_argmin_wmma<<<256 * 32, 256, 0, stream>>>(z4b, cbb, cnorm, keys);
  vq_finalize<<<8192, 256, 0, stream>>>(keys, cbf, z4f, qout, idxout, partials);
  reduce_loss<<<1, 256, 0, stream>>>(partials, lossp);
}
